// Linear_sp_28518582846030
// MI455X (gfx1250) — 8-bit matrix operand order repaired and checked by reading, compiled, not run on hardware
//
#include <hip/hip_runtime.h>
#include <hip/hip_bf16.h>

typedef __attribute__((ext_vector_type(8))) int v8i;

constexpr int M = 8192, N = 4096, K = 4096;
constexpr int KT = K / 64;
constexpr int NT = N / 16;
constexpr int MT = M / 16;
constexpr int NB = 4;

constexpr size_t OFF_MAX = 0;
constexpr size_t OFF_AS  = 256;
constexpr size_t SZ_AS   = (size_t)4 * M * K;
constexpr size_t OFF_BS  = OFF_AS + SZ_AS;
constexpr size_t SZ_BS   = (size_t)N * K;
constexpr size_t OFF_P   = OFF_BS + SZ_BS;
constexpr size_t SZ_P    = (size_t)4 * M * N * 4;
constexpr size_t OFF_Y   = OFF_P + SZ_P;

__global__ __launch_bounds__(32) void init_kernel(unsigned* maxbuf) {
    if (threadIdx.x < 8) maxbuf[threadIdx.x] = 0u;
}

__global__ __launch_bounds__(256) void absmax_kernel(const float* __restrict__ a, size_t n,
                                                     unsigned* __restrict__ slot) {
    float m = 0.0f;
    size_t stride = (size_t)gridDim.x * blockDim.x;
    for (size_t i = (size_t)blockIdx.x * blockDim.x + threadIdx.x; i < n; i += stride)
        m = fmaxf(m, fabsf(a[i]));
    #pragma unroll
    for (int off = 16; off > 0; off >>= 1)
        m = fmaxf(m, __shfl_xor(m, off, 32));
    __shared__ float smem[8];
    int lane = threadIdx.x & 31, w = threadIdx.x >> 5;
    if (lane == 0) smem[w] = m;
    __syncthreads();
    if (threadIdx.x == 0) {
        float mm = smem[0];
        #pragma unroll
        for (int i = 1; i < 8; ++i) mm = fmaxf(mm, smem[i]);
        atomicMax(slot, __float_as_uint(mm));
    }
}

__global__ __launch_bounds__(256) void quantize_x_kernel(const float* __restrict__ x,
                                                         unsigned char* __restrict__ As,
                                                         const unsigned* __restrict__ maxbuf) {
    float s = fmaxf(__uint_as_float(maxbuf[4]), 1e-12f) / 127.0f;
    float inv = 1.0f / s;
    size_t idx = (size_t)blockIdx.x * blockDim.x + threadIdx.x;
    size_t e = idx * 4;
    int m = (int)(e >> 12), k = (int)(e & (K - 1));
    const float4 xv = *(const float4*)(x + e);
    const float xs[4] = {xv.x, xv.y, xv.z, xv.w};
    unsigned pk[4] = {0u, 0u, 0u, 0u};
    #pragma unroll
    for (int b = 0; b < 4; ++b) {
        int iq = (int)rintf(xs[b] * inv);
        int sg = iq < 0 ? -1 : 1;
        int mag = iq < 0 ? -iq : iq;
        #pragma unroll
        for (int sl = 0; sl < 4; ++sl) {
            int v = ((mag >> (2 * sl)) & 3) * sg;
            pk[sl] |= ((unsigned)(v & 0xff)) << (8 * b);
        }
    }
    int mt = m >> 4, mm = m & 15, kt = k >> 6, kk = k & 63;
    int v = (((kk >> 5) & 1) << 2) | ((kk >> 2) & 3);
    int l = mm | (((kk >> 4) & 1) << 4);
    size_t off = (((size_t)(mt * KT + kt)) << 10) + (size_t)l * 32 + (size_t)v * 4;
    #pragma unroll
    for (int sl = 0; sl < 4; ++sl)
        *(unsigned*)(As + (size_t)sl * M * K + off) = pk[sl];
}

__global__ __launch_bounds__(256) void quantize_w_kernel(const float* __restrict__ w,
                                                         unsigned char* __restrict__ Bs,
                                                         const unsigned* __restrict__ maxbuf) {
    float s = fmaxf(__uint_as_float(maxbuf[5]), 1e-12f) / 127.0f;
    float inv = 1.0f / s;
    size_t idx = (size_t)blockIdx.x * blockDim.x + threadIdx.x;
    size_t e = idx * 4;
    int n = (int)(e >> 12), k = (int)(e & (K - 1));
    const float4 wv = *(const float4*)(w + e);
    const float ws[4] = {wv.x, wv.y, wv.z, wv.w};
    unsigned pk = 0u;
    #pragma unroll
    for (int b = 0; b < 4; ++b) {
        int iq = (int)rintf(ws[b] * inv);
        pk |= ((unsigned)(iq & 0xff)) << (8 * b);
    }
    int nt = n >> 4, nn = n & 15, kt = k >> 6, kk = k & 63;
    int v = (((kk >> 5) & 1) << 2) | ((kk >> 2) & 3);
    int l = nn | (((kk >> 4) & 1) << 4);
    *(unsigned*)(Bs + (((size_t)(nt * KT + kt)) << 10) + (size_t)l * 32 + (size_t)v * 4) = pk;
}

__global__ __launch_bounds__(256) void gemm_kernel(const unsigned char* __restrict__ As,
                                                   const unsigned char* __restrict__ Bs,
                                                   int* __restrict__ P,
                                                   unsigned* __restrict__ maxbuf) {
    int wave = (int)((blockIdx.x * blockDim.x + threadIdx.x) >> 5);
    wave = __builtin_amdgcn_readfirstlane(wave);
    int lane = threadIdx.x & 31;
    int mt  = wave % MT;
    int nt0 = (wave / MT) * NB;

    const unsigned char* aS[4];
    #pragma unroll
    for (int s = 0; s < 4; ++s)
        aS[s] = As + (size_t)s * M * K + (((size_t)mt * KT) << 10);
    const unsigned char* bS[NB];
    #pragma unroll
    for (int j = 0; j < NB; ++j)
        bS[j] = Bs + (((size_t)(nt0 + j) * KT) << 10);

    v8i acc[4][NB];
    #pragma unroll
    for (int s = 0; s < 4; ++s)
        #pragma unroll
        for (int j = 0; j < NB; ++j)
            acc[s][j] = (v8i){0, 0, 0, 0, 0, 0, 0, 0};

    int voff = lane * 32;
    for (int kt = 0; kt < KT; ++kt) {
        v8i a[4];
        #pragma unroll
        for (int s = 0; s < 4; ++s)
            a[s] = *(const v8i*)(aS[s] + voff);
        v8i b[NB];
        #pragma unroll
        for (int j = 0; j < NB; ++j)
            b[j] = *(const v8i*)(bS[j] + voff);
        #pragma unroll
        for (int s = 0; s < 4; ++s)
            #pragma unroll
            for (int j = 0; j < NB; ++j)
                acc[s][j] = __builtin_amdgcn_wmma_i32_16x16x64_iu8(
                     true, a[s],  true, b[j], acc[s][j],
                     false,  false);
        voff += 1024;
    }

    #pragma unroll
    for (int s = 0; s < 4; ++s) {
        int mymax = 0;
        #pragma unroll
        for (int j = 0; j < NB; ++j) {
            int* dst = P + (size_t)s * M * N + ((size_t)(mt * NT + nt0 + j)) * 256 + lane * 8;
            *(v8i*)dst = acc[s][j];
            #pragma unroll
            for (int v = 0; v < 8; ++v) {
                int av = acc[s][j][v];
                av = av < 0 ? -av : av;
                mymax = max(mymax, av);
            }
        }
        #pragma unroll
        for (int off = 16; off > 0; off >>= 1)
            mymax = max(mymax, __shfl_xor(mymax, off, 32));
        if (lane == 0) atomicMax(&maxbuf[s], (unsigned)mymax);
    }
}

__global__ __launch_bounds__(256) void combine_kernel(const int* __restrict__ P,
                                                      float* __restrict__ Y,
                                                      unsigned* __restrict__ maxbuf) {
    float sadc[4];
    #pragma unroll
    for (int s = 0; s < 4; ++s)
        sadc[s] = fmaxf((float)maxbuf[s], 1e-12f) / 127.0f;
    float sx = fmaxf(__uint_as_float(maxbuf[4]), 1e-12f) / 127.0f;
    float sw = fmaxf(__uint_as_float(maxbuf[5]), 1e-12f) / 127.0f;
    float sxw = sx * sw;
    const size_t MN = (size_t)M * N;
    float lmax = 0.0f;
    size_t stride = (size_t)gridDim.x * blockDim.x;
    for (size_t i = (size_t)blockIdx.x * blockDim.x + threadIdx.x; i < MN; i += stride) {
        float accv = 0.0f, p4 = 1.0f;
        #pragma unroll
        for (int s = 0; s < 4; ++s) {
            float p = (float)P[s * MN + i];
            p = rintf(p / sadc[s]) * sadc[s];
            accv += p * p4;
            p4 *= 4.0f;
        }
        float y = accv * sxw;
        Y[i] = y;
        lmax = fmaxf(lmax, fabsf(y));
    }
    #pragma unroll
    for (int off = 16; off > 0; off >>= 1)
        lmax = fmaxf(lmax, __shfl_xor(lmax, off, 32));
    if ((threadIdx.x & 31) == 0) atomicMax(&maxbuf[6], __float_as_uint(lmax));
}

__global__ __launch_bounds__(256) void final_kernel(const float* __restrict__ Y,
                                                    const float* __restrict__ bias,
                                                    float* __restrict__ out,
                                                    const unsigned* __restrict__ maxbuf) {
    float so = fmaxf(__uint_as_float(maxbuf[6]), 1e-12f) / 127.0f;
    const size_t MN = (size_t)M * N;
    size_t stride = (size_t)gridDim.x * blockDim.x;
    for (size_t i = (size_t)blockIdx.x * blockDim.x + threadIdx.x; i < MN; i += stride) {
        int mrow = (int)(i >> 12), n = (int)(i & (N - 1));
        int mt = mrow >> 4, mm = mrow & 15, nt = n >> 4, nn = n & 15;
        int v = mm & 7;
        int lane = nn + ((mm >> 3) << 4);
        size_t off = ((size_t)(mt * NT + nt)) * 256 + (size_t)lane * 8 + v;
        float y = Y[off];
        out[i] = rintf(y / so) * so + bias[n];
    }
}

extern "C" void kernel_launch(void* const* d_in, const int* in_sizes, int n_in,
                              void* d_out, int out_size, void* d_ws, size_t ws_size,
                              hipStream_t stream) {
    const float* x    = (const float*)d_in[0];
    const float* w    = (const float*)d_in[1];
    const float* bias = (const float*)d_in[2];
    float* out = (float*)d_out;

    unsigned char* ws = (unsigned char*)d_ws;
    unsigned* maxbuf  = (unsigned*)(ws + OFF_MAX);
    unsigned char* As = ws + OFF_AS;
    unsigned char* Bs = ws + OFF_BS;
    int*   P = (int*)(ws + OFF_P);
    float* Y = (float*)(ws + OFF_Y);

    init_kernel<<<1, 32, 0, stream>>>(maxbuf);

    absmax_kernel<<<2048, 256, 0, stream>>>(x, (size_t)M * K, &maxbuf[4]);
    absmax_kernel<<<2048, 256, 0, stream>>>(w, (size_t)N * K, &maxbuf[5]);

    quantize_x_kernel<<<(M * (K / 4)) / 256, 256, 0, stream>>>(x, As, maxbuf);
    quantize_w_kernel<<<(N * (K / 4)) / 256, 256, 0, stream>>>(w, Bs, maxbuf);

    gemm_kernel<<<(MT * (NT / NB)) / 8, 256, 0, stream>>>(As, Bs, P, maxbuf);

    combine_kernel<<<8192, 256, 0, stream>>>(P, Y, maxbuf);
    final_kernel<<<8192, 256, 0, stream>>>(Y, bias, out, maxbuf);
}
